// LinearBin_15066745274546
// MI455X (gfx1250) — matrix operand order repaired and checked by reading, not compiled, not run on hardware
//
#include <hip/hip_runtime.h>
#include <hip/hip_bf16.h>
#include <stdint.h>

// CDNA5 WMMA fragment types (wave32)
typedef __attribute__((ext_vector_type(16))) __bf16 v16bf;
typedef __attribute__((ext_vector_type(8)))  float  v8f;

#define TERN_THRESH 0.05f

__device__ __forceinline__ uint32_t f32_to_bf16_rne(float f) {
  uint32_t u = __float_as_uint(f);
  uint32_t r = u + 0x7FFFu + ((u >> 16) & 1u);
  return (r >> 16) & 0xFFFFu;
}
__device__ __forceinline__ float bf16_to_f32(uint32_t b) {
  return __uint_as_float(b << 16);
}

// ---------------------------------------------------------------------------
// Prep 1: ternarize W[4096(o) x 4096(i)] fp32 -> bf16 {-1,0,+1}, written in
// WMMA-B fragment-ready layout: dword[((kt*256+nt)*32+lane)*8+v] packs the
// bf16 pair (k0,k0+1) with n=nt*16+(lane&15), k0=kt*32+(lane>>4)*16+2v.
// (B 32x16 KxN layout; contiguous k order, matching the A buffer.)
// ---------------------------------------------------------------------------
__global__ __launch_bounds__(256) void quant_w_kernel(const float* __restrict__ w,
                                                      uint32_t* __restrict__ bw) {
  uint32_t tid = blockIdx.x * 256u + threadIdx.x;     // 8,388,608 dwords total
  uint32_t v  = tid & 7u;
  uint32_t l  = (tid >> 3) & 31u;
  uint32_t nt = (tid >> 8) & 255u;
  uint32_t kt = tid >> 16;
  uint32_t n  = nt * 16u + (l & 15u);
  uint32_t k0 = kt * 32u + (l >> 4) * 16u + v * 2u;
  float w0 = w[n * 4096u + k0];
  float w1 = w[n * 4096u + k0 + 1u];
  uint32_t t0 = (w0 > TERN_THRESH) ? 0x3F80u : ((w0 < -TERN_THRESH) ? 0xBF80u : 0u);
  uint32_t t1 = (w1 > TERN_THRESH) ? 0x3F80u : ((w1 < -TERN_THRESH) ? 0xBF80u : 0u);
  bw[tid] = t0 | (t1 << 16);
}

// ---------------------------------------------------------------------------
// Prep 2: split x fp32 -> bf16 hi + bf16 lo (x ~= hi+lo, ~16-bit mantissa),
// written in WMMA-A fragment-ready layout:
// dword[((mt*128+kt)*32+lane)*8+v] packs bf16 pair (k0,k0+1) with
// m=mt*16+(lane&15), k0=kt*32+(lane>>4)*16+2v.
// ---------------------------------------------------------------------------
__global__ __launch_bounds__(256) void split_x_kernel(const float* __restrict__ x,
                                                      uint32_t* __restrict__ ahi,
                                                      uint32_t* __restrict__ alo,
                                                      int write_lo) {
  uint32_t tid = blockIdx.x * 256u + threadIdx.x;     // 16,777,216 dwords total
  uint32_t v  = tid & 7u;
  uint32_t l  = (tid >> 3) & 31u;
  uint32_t kt = (tid >> 8) & 127u;
  uint32_t mt = tid >> 15;
  uint32_t m  = mt * 16u + (l & 15u);
  uint32_t k0 = kt * 32u + (l >> 4) * 16u + v * 2u;
  float x0 = x[m * 4096u + k0];
  float x1 = x[m * 4096u + k0 + 1u];
  uint32_t h0 = f32_to_bf16_rne(x0);
  uint32_t h1 = f32_to_bf16_rne(x1);
  ahi[tid] = h0 | (h1 << 16);
  if (write_lo) {
    uint32_t l0 = f32_to_bf16_rne(x0 - bf16_to_f32(h0));
    uint32_t l1 = f32_to_bf16_rne(x1 - bf16_to_f32(h1));
    alo[tid] = l0 | (l1 << 16);
  }
}

// ---------------------------------------------------------------------------
// GEMM: 256 threads (8 wave32) per block, 128x256 output tile per block,
// 64x64 per wave = 4x4 WMMA C-tiles (128 acc VGPRs), K-loop in steps of 32
// using v_wmma_f32_16x16x32_bf16. Fragments loaded straight from swizzled
// global buffers (L2-resident: 160 MiB < 192 MB L2); intra-block fragment
// redundancy is absorbed by the WGP$. Per K-step: 24 b128 loads : 32 WMMAs.
// ---------------------------------------------------------------------------
template <bool USE_LO>
__global__ __launch_bounds__(256) void gemm_kernel(const uint16_t* __restrict__ Ahi,
                                                   const uint16_t* __restrict__ Alo,
                                                   const uint16_t* __restrict__ Bw,
                                                   const float* __restrict__ bias,
                                                   float* __restrict__ out) {
  const uint32_t lane = threadIdx.x & 31u;
  const uint32_t wave = threadIdx.x >> 5;
  const uint32_t mt0 = blockIdx.x * 8u  + (wave & 1u) * 4u;  // 16-row tile index
  const uint32_t nt0 = blockIdx.y * 16u + (wave >> 1) * 4u;  // 16-col tile index

  v8f acc[4][4] = {};   // f32 accumulators, 128 VGPRs

#pragma unroll 1
  for (uint32_t kt = 0; kt < 128u; ++kt) {
    const uint16_t* bp = Bw + (((size_t)kt * 256u + nt0) * 32u + lane) * 16u;
    v16bf b0 = *(const v16bf*)(bp);
    v16bf b1 = *(const v16bf*)(bp + 512u);    // next n-tile: +32 lanes * 16 elems
    v16bf b2 = *(const v16bf*)(bp + 1024u);
    v16bf b3 = *(const v16bf*)(bp + 1536u);

#pragma unroll
    for (int im = 0; im < 4; ++im) {
      size_t aoff = (((size_t)(mt0 + (uint32_t)im) * 128u + kt) * 32u + lane) * 16u;
      v16bf ah = *(const v16bf*)(Ahi + aoff);
      acc[im][0] = __builtin_amdgcn_wmma_f32_16x16x32_bf16(
          false, ah, false, b0, (short)0, acc[im][0], false, false);
      acc[im][1] = __builtin_amdgcn_wmma_f32_16x16x32_bf16(
          false, ah, false, b1, (short)0, acc[im][1], false, false);
      acc[im][2] = __builtin_amdgcn_wmma_f32_16x16x32_bf16(
          false, ah, false, b2, (short)0, acc[im][2], false, false);
      acc[im][3] = __builtin_amdgcn_wmma_f32_16x16x32_bf16(
          false, ah, false, b3, (short)0, acc[im][3], false, false);
      if (USE_LO) {
        v16bf al = *(const v16bf*)(Alo + aoff);
        acc[im][0] = __builtin_amdgcn_wmma_f32_16x16x32_bf16(
            false, al, false, b0, (short)0, acc[im][0], false, false);
        acc[im][1] = __builtin_amdgcn_wmma_f32_16x16x32_bf16(
            false, al, false, b1, (short)0, acc[im][1], false, false);
        acc[im][2] = __builtin_amdgcn_wmma_f32_16x16x32_bf16(
            false, al, false, b2, (short)0, acc[im][2], false, false);
        acc[im][3] = __builtin_amdgcn_wmma_f32_16x16x32_bf16(
            false, al, false, b3, (short)0, acc[im][3], false, false);
      }
    }
  }

  // Epilogue: C/D 16x16 f32 layout -> n = lane&15, m = vgpr + (lane>>4)*8
  const uint32_t nl = lane & 15u;
  const uint32_t mh = (lane >> 4) * 8u;
#pragma unroll
  for (int im = 0; im < 4; ++im) {
    uint32_t mb = (mt0 + (uint32_t)im) * 16u + mh;
#pragma unroll
    for (int in = 0; in < 4; ++in) {
      uint32_t n = (nt0 + (uint32_t)in) * 16u + nl;
      float bv = bias[n];
#pragma unroll
      for (int r = 0; r < 8; ++r) {
        out[(size_t)(mb + (uint32_t)r) * 4096u + n] = acc[im][in][r] + bv;
      }
    }
  }
}

extern "C" void kernel_launch(void* const* d_in, const int* in_sizes, int n_in,
                              void* d_out, int out_size, void* d_ws, size_t ws_size,
                              hipStream_t stream) {
  (void)in_sizes; (void)n_in; (void)out_size;
  const float* x    = (const float*)d_in[0];   // [8192,4096] fp32
  const float* w    = (const float*)d_in[1];   // [4096,4096] fp32
  const float* bias = (const float*)d_in[2];   // [4096] fp32
  float* out = (float*)d_out;                  // [8192,4096] fp32

  char* ws = (char*)d_ws;
  const size_t SZ_B = 32ull << 20;   // swizzled ternary W (bf16)
  const size_t SZ_A = 64ull << 20;   // each swizzled x term (bf16)
  uint32_t* Bw  = (uint32_t*)ws;
  uint32_t* Ahi = (uint32_t*)(ws + SZ_B);
  uint32_t* Alo = (uint32_t*)(ws + SZ_B + SZ_A);
  const bool use_lo = ws_size >= (SZ_B + 2u * SZ_A);

  quant_w_kernel<<<32768, 256, 0, stream>>>(w, Bw);
  split_x_kernel<<<65536, 256, 0, stream>>>(x, Ahi, use_lo ? Alo : Ahi,
                                            use_lo ? 1 : 0);

  dim3 grid(64, 16);   // 8192/128 x 4096/256
  if (use_lo) {
    gemm_kernel<true><<<grid, 256, 0, stream>>>(
        (const uint16_t*)Ahi, (const uint16_t*)Alo, (const uint16_t*)Bw, bias, out);
  } else {
    gemm_kernel<false><<<grid, 256, 0, stream>>>(
        (const uint16_t*)Ahi, (const uint16_t*)Ahi, (const uint16_t*)Bw, bias, out);
  }
}
